// EncoderLayer_45715631898866
// MI455X (gfx1250) — matrix operand order repaired and checked by reading, not compiled, not run on hardware
//
#include <hip/hip_runtime.h>
#include <hip/hip_bf16.h>

// ---------------- problem constants ----------------
constexpr int kB  = 4;
constexpr int kT  = 2048;
constexpr int kBT = kB * kT;      // 8192 tokens
constexpr int kD  = 1024;
constexpr int kH  = 8;
constexpr int kHD = 128;
constexpr int kE  = 8;
constexpr int kF  = 4096;         // FFN hidden

typedef __bf16 bf16;
typedef __attribute__((ext_vector_type(16))) __bf16 v16bf;
typedef __attribute__((ext_vector_type(8)))  __bf16 v8bf;
typedef __attribute__((ext_vector_type(8)))  float  v8f;
typedef __attribute__((ext_vector_type(4)))  unsigned int v4u;
typedef __attribute__((ext_vector_type(8)))  int v8i;
typedef __attribute__((ext_vector_type(4)))  int v4i;

union V16 { v16bf v; v8bf h[2]; };

#if defined(__has_builtin)
#if __has_builtin(__builtin_amdgcn_tensor_load_to_lds) && \
    __has_builtin(__builtin_amdgcn_s_wait_tensorcnt)
#define HAVE_TDM 1
#endif
#endif

__device__ __forceinline__ v8f wmma_bf16(const v16bf& a, const v16bf& b, const v8f& c) {
  // D = A(16x32 bf16) x B(32x16 bf16) + C(16x16 f32)
  return __builtin_amdgcn_wmma_f32_16x16x32_bf16(false, a, false, b, (short)0, c,
                                                 false, false);
}

// padded LDS pitch (in bf16 elements): 40 halfs = 80B -> conflict-free fragment rows
constexpr int kPitch = 40;

#ifdef HAVE_TDM
__device__ __forceinline__ unsigned lds_addr_of(const void* p) {
  // generic LDS pointer: low 32 bits are the LDS byte offset (ISA 10.2 aperture rules)
  return (unsigned)(unsigned long long)(size_t)p;
}

// 2D bf16 tile load via Tensor Data Mover (ISA ch.8 D# layout), with LDS padding:
// every 16 dwords (one 32-half row) insert 4 dwords -> row pitch 40 halfs.
// This toolchain exposes the 6-arg builtin: (v4u g0, v8i g1, v4i g2, v4i g3, v8i, i32 cpol)
__device__ __forceinline__ void tdm_load_a_tile(unsigned lds_byte_addr,
                                                const bf16* gptr, unsigned strideElems) {
  const unsigned long long ga = (unsigned long long)(size_t)gptr;
  v4u g0;
  g0[0] = 1u;                                   // count=1, user descriptor
  g0[1] = lds_byte_addr;                        // lds_addr
  g0[2] = (unsigned)(ga & 0xffffffffu);         // global_addr[31:0]
  g0[3] = (unsigned)((ga >> 32) & 0x01ffffffu)  // global_addr[56:32]
        | (2u << 30);                           // type = 2 ("image")
  v8i g1;
  g1[0] = (int)((1u << 16)                      // data_size = 1 (2 bytes)
        |       (1u << 20)                      // pad_enable
        |       (3u << 22)                      // pad_interval: 16 dwords
        |       (3u << 25));                    // pad_amount:   4 dwords
  g1[1] = (int)(32u << 16);                     // tensor_dim0 = 32 (bits 79:48 lo)
  g1[2] = (int)(128u << 16);                    // tensor_dim0 hi=0 | tensor_dim1 = 128
  g1[3] = (int)(32u << 16);                     // tensor_dim1 hi=0 | tile_dim0 = 32
  g1[4] = (int)128;                             // tile_dim1 = 128, tile_dim2 = 0
  g1[5] = (int)strideElems;                     // tensor_dim0_stride (elements)
  g1[6] = 0;
  g1[7] = 0;
  const v4i z4 = {0, 0, 0, 0};
  const v8i z8 = {0, 0, 0, 0, 0, 0, 0, 0};
  __builtin_amdgcn_tensor_load_to_lds(g0, g1, z4, z4, z8, 0);
}
#endif

// ---------------- f32 -> bf16 convert ----------------
__global__ void cvt_kernel(const float* __restrict__ in, bf16* __restrict__ out, size_t n) {
  size_t i = (size_t)blockIdx.x * blockDim.x + threadIdx.x;
  size_t stride = (size_t)gridDim.x * blockDim.x;
  for (; i < n; i += stride) out[i] = (bf16)in[i];
}

// ---------------- generic WMMA GEMM ----------------
// C[M,N] = op(A[M,K]_bf16 @ B[K,N]_bf16 + bias)   (M%128==0, N%64==0, K%32==0)
// flags: 1 = relu, 2 = accumulate into Cf
__global__ __launch_bounds__(256) void gemm_bf16_kernel(
    const bf16* __restrict__ A, const bf16* __restrict__ B,
    const float* __restrict__ bias,
    float* __restrict__ Cf, bf16* __restrict__ Cb,
    const float* __restrict__ rowscale, int rs_stride,
    int M, int N, int K, int flags) {
  __shared__ bf16 As[2 * 128 * kPitch];  // double-buffered A tile [m][k], pitch 40
  __shared__ bf16 Bt[64 * kPitch];       // B tile transposed [n][k], pitch 40

  const int tid  = threadIdx.x;
  const int wave = tid >> 5;
  const int lane = tid & 31;
  const int g    = lane >> 4;     // lane group 0/1
  const int ln   = lane & 15;
  const int rowBase = blockIdx.y * 128;
  const int n0      = blockIdx.x * 64;

  v8f acc[4] = {};

  const int bk = tid >> 3;            // 0..31
  const int bn = (tid & 7) * 8;       // 0..56
  const int nk = K / 32;

#ifdef HAVE_TDM
  const bf16* Atile = A + (size_t)rowBase * K;
  if (wave == 0) tdm_load_a_tile(lds_addr_of(&As[0]), Atile, (unsigned)K);
  int cur = 0;
  for (int ik = 0; ik < nk; ++ik) {
    // stage B tile transposed (all threads)
    v8bf bv = *(const v8bf*)(B + (size_t)(ik * 32 + bk) * N + n0 + bn);
#pragma unroll
    for (int j = 0; j < 8; ++j) Bt[(bn + j) * kPitch + bk] = bv[j];
    if (ik + 1 < nk)
      __builtin_prefetch(B + (size_t)((ik + 1) * 32 + bk) * N + n0 + bn, 0, 3);
    if (wave == 0) {
      if (ik + 1 < nk) {  // pipeline: issue next A tile, wait for current one only
        tdm_load_a_tile(lds_addr_of(&As[(cur ^ 1) * 128 * kPitch]),
                        Atile + (size_t)(ik + 1) * 32, (unsigned)K);
        __builtin_amdgcn_s_wait_tensorcnt(1);
      } else {
        __builtin_amdgcn_s_wait_tensorcnt(0);
      }
    }
    __syncthreads();
    const bf16* Ac = &As[cur * 128 * kPitch];
    V16 a;  // A frag: lanes0-15 K{0..7,16..23}, lanes16-31 K{8..15,24..31}
    a.h[0] = *(const v8bf*)&Ac[(wave * 16 + ln) * kPitch + g * 8];
    a.h[1] = *(const v8bf*)&Ac[(wave * 16 + ln) * kPitch + 16 + g * 8];
#pragma unroll
    for (int nt = 0; nt < 4; ++nt) {
      V16 b;  // B frag: lane column ln, K{g*8..g*8+7, 16+g*8..16+g*8+7}
      b.h[0] = *(const v8bf*)&Bt[(nt * 16 + ln) * kPitch + g * 8];
      b.h[1] = *(const v8bf*)&Bt[(nt * 16 + ln) * kPitch + 16 + g * 8];
      acc[nt] = wmma_bf16(a.v, b.v, acc[nt]);
    }
    __syncthreads();
    cur ^= 1;
  }
#else
  const int arow = tid >> 1;            // 0..127
  const int acol = (tid & 1) * 16;      // 0 or 16
  for (int ik = 0; ik < nk; ++ik) {
    const bf16* ap = A + (size_t)(rowBase + arow) * K + ik * 32 + acol;
    *(v8bf*)&As[arow * kPitch + acol]     = *(const v8bf*)(ap);
    *(v8bf*)&As[arow * kPitch + acol + 8] = *(const v8bf*)(ap + 8);
    v8bf bv = *(const v8bf*)(B + (size_t)(ik * 32 + bk) * N + n0 + bn);
#pragma unroll
    for (int j = 0; j < 8; ++j) Bt[(bn + j) * kPitch + bk] = bv[j];
    if (ik + 1 < nk)
      __builtin_prefetch(B + (size_t)((ik + 1) * 32 + bk) * N + n0 + bn, 0, 3);
    __syncthreads();
    V16 a;
    a.h[0] = *(const v8bf*)&As[(wave * 16 + ln) * kPitch + g * 8];
    a.h[1] = *(const v8bf*)&As[(wave * 16 + ln) * kPitch + 16 + g * 8];
#pragma unroll
    for (int nt = 0; nt < 4; ++nt) {
      V16 b;
      b.h[0] = *(const v8bf*)&Bt[(nt * 16 + ln) * kPitch + g * 8];
      b.h[1] = *(const v8bf*)&Bt[(nt * 16 + ln) * kPitch + 16 + g * 8];
      acc[nt] = wmma_bf16(a.v, b.v, acc[nt]);
    }
    __syncthreads();
  }
#endif

  // epilogue: C/D layout: VGPR r -> M = r + 8*g, N = ln
#pragma unroll
  for (int nt = 0; nt < 4; ++nt) {
    const int col = n0 + nt * 16 + ln;
    const float bs = bias ? bias[col] : 0.f;
#pragma unroll
    for (int r = 0; r < 8; ++r) {
      const int grow = rowBase + wave * 16 + r + 8 * g;
      float v = acc[nt][r] + bs;
      if (flags & 1) v = fmaxf(v, 0.f);
      if (rowscale) v *= rowscale[(size_t)grow * rs_stride];
      const size_t o = (size_t)grow * N + col;
      if (Cf) { if (flags & 2) Cf[o] += v; else Cf[o] = v; }
      if (Cb) Cb[o] = (bf16)v;
    }
  }
}

// ---------------- attention (full score slab in 320KB LDS) ----------------
// grid: (T/32, B*H). block: 256 threads. Q/K/V/O: bf16 [B*T, D], head h at col h*HD.
__global__ __launch_bounds__(256) void attn_kernel(
    const bf16* __restrict__ Q, const bf16* __restrict__ Kd,
    const bf16* __restrict__ V, bf16* __restrict__ O) {
  extern __shared__ char smem[];
  float* S  = (float*)smem;                                        // 32 x 2048 f32
  bf16*  Qs = (bf16*)(smem + 32 * 2048 * 4);                       // 32 x 128
  bf16*  Ks = (bf16*)(smem + 32 * 2048 * 4 + 32 * 128 * 2);        // 64 x 128 (reused as Vt 128x32)
  float* rs = (float*)(smem + 32 * 2048 * 4 + 32 * 128 * 2 + 64 * 128 * 2);  // 32

  const int tid  = threadIdx.x;
  const int wave = tid >> 5;
  const int lane = tid & 31;
  const int g    = lane >> 4;
  const int ln   = lane & 15;
  const int q0   = blockIdx.x * 32;
  const int h    = blockIdx.y % kH;
  const int b    = blockIdx.y / kH;
  const size_t base = (size_t)b * kT * kD + (size_t)h * kHD;

  // load Q tile 32x128
  {
    const int row = tid >> 3, c0 = (tid & 7) * 16;
    const bf16* qp = Q + base + (size_t)(q0 + row) * kD + c0;
    *(v8bf*)&Qs[row * 128 + c0]     = *(const v8bf*)qp;
    *(v8bf*)&Qs[row * 128 + c0 + 8] = *(const v8bf*)(qp + 8);
  }

  const int qt = wave & 1;   // query sub-tile
  const int kt = wave >> 1;  // key sub-tile within 64-key chunk
  const float scale = 0.08838834764831845f;  // 1/sqrt(128)

  // ---- phase 1: S = (Q @ K^T) * scale ----
  for (int kc = 0; kc < kT; kc += 64) {
    const int row = tid >> 2, c0 = (tid & 3) * 32;
    const bf16* kp = Kd + base + (size_t)(kc + row) * kD + c0;
    *(v8bf*)&Ks[row * 128 + c0]      = *(const v8bf*)kp;
    *(v8bf*)&Ks[row * 128 + c0 + 8]  = *(const v8bf*)(kp + 8);
    *(v8bf*)&Ks[row * 128 + c0 + 16] = *(const v8bf*)(kp + 16);
    *(v8bf*)&Ks[row * 128 + c0 + 24] = *(const v8bf*)(kp + 24);
    __syncthreads();

    v8f acc = {};
#pragma unroll
    for (int kkk = 0; kkk < 4; ++kkk) {  // HD=128 = 4 * K32
      V16 a, bb;
      a.h[0]  = *(const v8bf*)&Qs[(qt * 16 + ln) * 128 + kkk * 32 + g * 8];
      a.h[1]  = *(const v8bf*)&Qs[(qt * 16 + ln) * 128 + kkk * 32 + 16 + g * 8];
      // B = K^T: Bt[n][k] == Ks[key=n][d=k]; row-major K tile is already transposed-staged
      bb.h[0] = *(const v8bf*)&Ks[(kt * 16 + ln) * 128 + kkk * 32 + g * 8];
      bb.h[1] = *(const v8bf*)&Ks[(kt * 16 + ln) * 128 + kkk * 32 + 16 + g * 8];
      acc = wmma_bf16(a.v, bb.v, acc);
    }
#pragma unroll
    for (int r = 0; r < 8; ++r)
      S[(qt * 16 + r + 8 * g) * 2048 + kc + kt * 16 + ln] = acc[r] * scale;
    __syncthreads();
  }

  // ---- phase 2: softmax rows (8 threads per row, stride-8 to spread LDS banks) ----
  {
    const int row = tid >> 3, j = tid & 7;
    float m = -3.4e38f;
    for (int c = j; c < 2048; c += 8) m = fmaxf(m, S[row * 2048 + c]);
    m = fmaxf(m, __shfl_xor(m, 1, 32));
    m = fmaxf(m, __shfl_xor(m, 2, 32));
    m = fmaxf(m, __shfl_xor(m, 4, 32));
    float s = 0.f;
    for (int c = j; c < 2048; c += 8) {
      float p = __expf(S[row * 2048 + c] - m);
      S[row * 2048 + c] = p;
      s += p;
    }
    s += __shfl_xor(s, 1, 32);
    s += __shfl_xor(s, 2, 32);
    s += __shfl_xor(s, 4, 32);
    if (j == 0) rs[row] = s;
  }
  __syncthreads();

  // ---- phase 3: O = P @ V (normalization folded into epilogue) ----
  v8f oacc[2] = {};
  const int nt0 = (wave >> 1) * 2;  // each wave owns 2 of 8 HD col-tiles
  bf16* Vt = Ks;                    // reuse as Vt[d][key] 128x32
  for (int kc = 0; kc < kT; kc += 32) {
    {
      const int key = tid & 31, d0 = (tid >> 5) * 16;
      const bf16* vp = V + base + (size_t)(kc + key) * kD + d0;
      v8bf x0 = *(const v8bf*)vp;
      v8bf x1 = *(const v8bf*)(vp + 8);
#pragma unroll
      for (int j = 0; j < 8; ++j) {
        Vt[(d0 + j) * 32 + key]     = x0[j];
        Vt[(d0 + 8 + j) * 32 + key] = x1[j];
      }
    }
    __syncthreads();
    V16 a;
#pragma unroll
    for (int i = 0; i < 8; ++i) {
      a.v[i]     = (bf16)S[(qt * 16 + ln) * 2048 + kc + g * 8 + i];
      a.v[i + 8] = (bf16)S[(qt * 16 + ln) * 2048 + kc + 16 + g * 8 + i];
    }
#pragma unroll
    for (int u = 0; u < 2; ++u) {
      const int nt = nt0 + u;
      V16 bb;
      bb.h[0] = *(const v8bf*)&Vt[(nt * 16 + ln) * 32 + g * 8];
      bb.h[1] = *(const v8bf*)&Vt[(nt * 16 + ln) * 32 + 16 + g * 8];
      oacc[u] = wmma_bf16(a.v, bb.v, oacc[u]);
    }
    __syncthreads();
  }
#pragma unroll
  for (int u = 0; u < 2; ++u) {
    const int nt = nt0 + u;
#pragma unroll
    for (int r = 0; r < 8; ++r) {
      const int rq = qt * 16 + r + 8 * g;
      const float v = oacc[u][r] / rs[rq];
      O[base + (size_t)(q0 + rq) * kD + nt * 16 + ln] = (bf16)v;
    }
  }
}

// ---------------- residual + LayerNorm ----------------
__global__ __launch_bounds__(256) void ln_kernel(
    const float* __restrict__ X, const float* __restrict__ R,
    const float* __restrict__ gw, const float* __restrict__ bw,
    float* __restrict__ Of, bf16* __restrict__ Ob) {
  __shared__ float red[256];
  const int t = blockIdx.x, tid = threadIdx.x;
  float vals[4];
  float s = 0.f;
#pragma unroll
  for (int i = 0; i < 4; ++i) {
    const int d = tid + i * 256;
    const float v = X[(size_t)t * kD + d] + R[(size_t)t * kD + d];
    vals[i] = v;
    s += v;
  }
  red[tid] = s;
  __syncthreads();
  for (int o = 128; o; o >>= 1) { if (tid < o) red[tid] += red[tid + o]; __syncthreads(); }
  const float mu = red[0] / kD;
  __syncthreads();
  float s2 = 0.f;
#pragma unroll
  for (int i = 0; i < 4; ++i) { const float dd = vals[i] - mu; s2 += dd * dd; }
  red[tid] = s2;
  __syncthreads();
  for (int o = 128; o; o >>= 1) { if (tid < o) red[tid] += red[tid + o]; __syncthreads(); }
  const float inv = rsqrtf(red[0] / kD + 1e-5f);
#pragma unroll
  for (int i = 0; i < 4; ++i) {
    const int d = tid + i * 256;
    const float y = (vals[i] - mu) * inv * gw[d] + bw[d];
    if (Of) Of[(size_t)t * kD + d] = y;
    if (Ob) Ob[(size_t)t * kD + d] = (bf16)y;
  }
}

// ---------------- noisy top-2 router ----------------
__global__ __launch_bounds__(128) void router_kernel(
    const float* __restrict__ Hs, const float* __restrict__ rw, const float* __restrict__ rb,
    const float* __restrict__ nw, const float* __restrict__ nb,
    const float* __restrict__ noise, float* __restrict__ probs) {
  __shared__ float red[16][128];
  const int t = blockIdx.x, tid = threadIdx.x;
  float acc[16] = {};
  for (int d = tid; d < kD; d += 128) {
    const float hv = Hs[(size_t)t * kD + d];
#pragma unroll
    for (int e = 0; e < 8; ++e) {
      acc[e]     += hv * rw[d * 8 + e];
      acc[8 + e] += hv * nw[d * 8 + e];
    }
  }
#pragma unroll
  for (int e = 0; e < 16; ++e) red[e][tid] = acc[e];
  __syncthreads();
  for (int o = 64; o; o >>= 1) {
    if (tid < o)
#pragma unroll
      for (int e = 0; e < 16; ++e) red[e][tid] += red[e][tid + o];
    __syncthreads();
  }
  if (tid == 0) {
    float noisy[8];
#pragma unroll
    for (int e = 0; e < 8; ++e) {
      const float lg = red[e][0] + rb[e];
      const float nl = red[8 + e][0] + nb[e];
      const float sp = (nl > 20.f) ? nl : log1pf(__expf(nl));  // softplus
      noisy[e] = lg + noise[(size_t)t * 8 + e] * sp;
    }
    int i1 = 0;
#pragma unroll
    for (int e = 1; e < 8; ++e) if (noisy[e] > noisy[i1]) i1 = e;
    int i2 = (i1 == 0) ? 1 : 0;
#pragma unroll
    for (int e = 0; e < 8; ++e) if (e != i1 && noisy[e] > noisy[i2]) i2 = e;
    const float m  = fmaxf(noisy[i1], noisy[i2]);
    const float e1 = __expf(noisy[i1] - m), e2 = __expf(noisy[i2] - m);
    const float inv = 1.f / (e1 + e2);
#pragma unroll
    for (int e = 0; e < 8; ++e) probs[(size_t)t * 8 + e] = 0.f;
    probs[(size_t)t * 8 + i1] = e1 * inv;
    probs[(size_t)t * 8 + i2] = e2 * inv;
  }
}

// ---------------- load-balance loss (deterministic single-block) ----------------
__global__ __launch_bounds__(256) void lb_kernel(const float* __restrict__ probs,
                                                 float* __restrict__ out) {
  __shared__ float red[16][256];
  const int tid = threadIdx.x;
  float sp[8] = {}, sc[8] = {};
  for (int t = tid; t < kBT; t += 256)
#pragma unroll
    for (int e = 0; e < 8; ++e) {
      const float p = probs[(size_t)t * 8 + e];
      sp[e] += p;
      sc[e] += (p > 0.f) ? 1.f : 0.f;
    }
#pragma unroll
  for (int e = 0; e < 8; ++e) { red[e][tid] = sp[e]; red[8 + e][tid] = sc[e]; }
  __syncthreads();
  for (int o = 128; o; o >>= 1) {
    if (tid < o)
#pragma unroll
      for (int e = 0; e < 16; ++e) red[e][tid] += red[e][tid + o];
    __syncthreads();
  }
  if (tid == 0) {
    float l = 0.f;
#pragma unroll
    for (int e = 0; e < 8; ++e)
      l += (red[e][0] / kBT) * (red[8 + e][0] / kBT);
    out[0] = (float)kE * l;
  }
}

// ---------------- host orchestration ----------------
static void launch_cvt(const float* in, bf16* out, size_t n, hipStream_t s) {
  const int blocks = (int)((n + 255) / 256 < 4096 ? (n + 255) / 256 : 4096);
  cvt_kernel<<<blocks, 256, 0, s>>>(in, out, n);
}

static void launch_gemm(const bf16* A, const bf16* B, const float* bias,
                        float* Cf, bf16* Cb, const float* rowscale, int rs_stride,
                        int M, int N, int K, int flags, hipStream_t s) {
  dim3 grid(N / 64, M / 128);
  gemm_bf16_kernel<<<grid, 256, 0, s>>>(A, B, bias, Cf, Cb, rowscale, rs_stride,
                                        M, N, K, flags);
}

extern "C" void kernel_launch(void* const* d_in, const int* in_sizes, int n_in,
                              void* d_out, int out_size, void* d_ws, size_t ws_size,
                              hipStream_t stream) {
  const float* x     = (const float*)d_in[0];
  const float* noise = (const float*)d_in[1];
  const float* wq = (const float*)d_in[2];  const float* bq = (const float*)d_in[3];
  const float* wk = (const float*)d_in[4];  const float* bk = (const float*)d_in[5];
  const float* wv = (const float*)d_in[6];  const float* bv = (const float*)d_in[7];
  const float* wo = (const float*)d_in[8];  const float* bo = (const float*)d_in[9];
  const float* g1 = (const float*)d_in[10]; const float* be1 = (const float*)d_in[11];
  const float* rw = (const float*)d_in[12]; const float* rb = (const float*)d_in[13];
  const float* nw = (const float*)d_in[14]; const float* nb = (const float*)d_in[15];
  const float* ew1 = (const float*)d_in[16]; const float* eb1 = (const float*)d_in[17];
  const float* ew2 = (const float*)d_in[18]; const float* eb2 = (const float*)d_in[19];
  const float* g2 = (const float*)d_in[20]; const float* be2 = (const float*)d_in[21];
  float* outF = (float*)d_out;  // [BT*D] y, then [1] lb_loss

  char* ws = (char*)d_ws;
  size_t off = 0;
  auto alloc = [&](size_t bytes) -> char* {
    char* p = ws + off;
    off += (bytes + 255) & ~(size_t)255;
    return p;
  };
  bf16*  wq_b  = (bf16*)alloc((size_t)kD * kD * 2);
  bf16*  wk_b  = (bf16*)alloc((size_t)kD * kD * 2);
  bf16*  wv_b  = (bf16*)alloc((size_t)kD * kD * 2);
  bf16*  wo_b  = (bf16*)alloc((size_t)kD * kD * 2);
  bf16*  ew1_b = (bf16*)alloc((size_t)kE * kD * kF * 2);  // 67 MB — L2-resident with ew2
  bf16*  ew2_b = (bf16*)alloc((size_t)kE * kF * kD * 2);
  bf16*  xb    = (bf16*)alloc((size_t)kBT * kD * 2);
  bf16*  qb    = (bf16*)alloc((size_t)kBT * kD * 2);
  bf16*  kb    = (bf16*)alloc((size_t)kBT * kD * 2);
  bf16*  vb    = (bf16*)alloc((size_t)kBT * kD * 2);
  bf16*  ao_b  = (bf16*)alloc((size_t)kBT * kD * 2);
  float* oproj = (float*)alloc((size_t)kBT * kD * 4);
  float* hbuf  = (float*)alloc((size_t)kBT * kD * 4);
  bf16*  hb    = (bf16*)alloc((size_t)kBT * kD * 2);
  float* probs = (float*)alloc((size_t)kBT * kE * 4);
  bf16*  h1    = (bf16*)alloc((size_t)kBT * kF * 2);
  float* facc  = (float*)alloc((size_t)kBT * kD * 4);
  (void)ws_size; (void)in_sizes; (void)n_in; (void)out_size;

  // 0) bf16 conversions
  launch_cvt(wq, wq_b, (size_t)kD * kD, stream);
  launch_cvt(wk, wk_b, (size_t)kD * kD, stream);
  launch_cvt(wv, wv_b, (size_t)kD * kD, stream);
  launch_cvt(wo, wo_b, (size_t)kD * kD, stream);
  launch_cvt(ew1, ew1_b, (size_t)kE * kD * kF, stream);
  launch_cvt(ew2, ew2_b, (size_t)kE * kF * kD, stream);
  launch_cvt(x, xb, (size_t)kBT * kD, stream);

  // 1) QKV projections (bf16 out for attention)
  launch_gemm(xb, wq_b, bq, nullptr, qb, nullptr, 1, kBT, kD, kD, 0, stream);
  launch_gemm(xb, wk_b, bk, nullptr, kb, nullptr, 1, kBT, kD, kD, 0, stream);
  launch_gemm(xb, wv_b, bv, nullptr, vb, nullptr, 1, kBT, kD, kD, 0, stream);

  // 2) attention (280KB dynamic LDS — CDNA5 WGP has 320KB)
  const int attn_smem = 32 * 2048 * 4 + 32 * 128 * 2 + 64 * 128 * 2 + 32 * 4;
  (void)hipFuncSetAttribute((const void*)attn_kernel,
                            hipFuncAttributeMaxDynamicSharedMemorySize, attn_smem);
  attn_kernel<<<dim3(kT / 32, kB * kH), 256, attn_smem, stream>>>(qb, kb, vb, ao_b);

  // 3) output projection + residual LayerNorm 1
  launch_gemm(ao_b, wo_b, bo, oproj, nullptr, nullptr, 1, kBT, kD, kD, 0, stream);
  ln_kernel<<<kBT, 256, 0, stream>>>(x, oproj, g1, be1, hbuf, hb);

  // 4) router + aux loss
  router_kernel<<<kBT, 128, 0, stream>>>(hbuf, rw, rb, nw, nb, noise, probs);
  lb_kernel<<<1, 256, 0, stream>>>(probs, outF + (size_t)kBT * kD);

  // 5) expert FFNs: out += prob_e * (relu(h@W1e+b1e)@W2e + b2e)
  (void)hipMemsetAsync(facc, 0, (size_t)kBT * kD * 4, stream);
  for (int e = 0; e < kE; ++e) {
    launch_gemm(hb, ew1_b + (size_t)e * kD * kF, eb1 + (size_t)e * kF,
                nullptr, h1, nullptr, 1, kBT, kF, kD, /*relu*/ 1, stream);
    launch_gemm(h1, ew2_b + (size_t)e * kF * kD, eb2 + (size_t)e * kD,
                facc, nullptr, probs + e, kE, kBT, kD, kF, /*accum*/ 2, stream);
  }

  // 6) final residual LayerNorm -> y
  ln_kernel<<<kBT, 256, 0, stream>>>(hbuf, facc, g2, be2, outF, nullptr);
}
